// PDENet_11158325035390
// MI455X (gfx1250) — matrix operand order repaired and checked by reading, not compiled, not run on hardware
//
#include <hip/hip_runtime.h>
#include <hip/hip_bf16.h>

typedef __attribute__((ext_vector_type(16))) _Float16 vh16;
typedef __attribute__((ext_vector_type(8)))  _Float16 vh8;
typedef __attribute__((ext_vector_type(8)))  float    vf8;

#define H 192
#define W 192
#define HW (192*192)
#define PW 194                 // padded width/height (1-pixel halo)
#define CAT16(lo,hi) __builtin_shufflevector((lo),(hi),0,1,2,3,4,5,6,7,8,9,10,11,12,13,14,15)

// ---------------- generic zero fill (dwords) ----------------
__global__ void zero_fill(uint32_t* __restrict__ p, int n) {
    int i = blockIdx.x * 256 + threadIdx.x;
    if (i < n) p[i] = 0u;
}

// ---------------- weight packing: OIHW fp32 -> [tap][O][I] f16 ----------------
__global__ void pack_w_enc2(const float* __restrict__ w, _Float16* __restrict__ wp) {
    int i = blockIdx.x * 256 + threadIdx.x;           // 9*128*64
    if (i >= 9 * 128 * 64) return;
    int c = i & 63; int k = (i >> 6) & 127; int t = i >> 13;
    wp[(t * 128 + k) * 64 + c] = (_Float16)w[((k * 64 + c) * 9) + t];
}
__global__ void pack_w_dec1(const float* __restrict__ w, _Float16* __restrict__ wp) {
    int i = blockIdx.x * 256 + threadIdx.x;           // 9*64*128
    if (i >= 9 * 64 * 128) return;
    int c = i & 127; int k = (i >> 7) & 63; int t = i >> 13;
    wp[(t * 64 + k) * 128 + c] = (_Float16)w[((k * 128 + c) * 9) + t];
}

// ---------------- enc1: conv 3->64 + relu, write PADDED NHWC f16 ----------------
__global__ void enc1_kernel(const float* __restrict__ x, const float* __restrict__ w1,
                            const float* __restrict__ b1, _Float16* __restrict__ hP) {
    int idx = blockIdx.x * 256 + threadIdx.x;         // HW*64
    if (idx >= HW * 64) return;
    int k = idx & 63; int p = idx >> 6;
    int y = p / W, xx = p % W;
    float acc = b1[k];
    #pragma unroll
    for (int c = 0; c < 3; ++c)
        #pragma unroll
        for (int dy = -1; dy <= 1; ++dy)
            #pragma unroll
            for (int dx = -1; dx <= 1; ++dx) {
                int yy = y + dy, x2 = xx + dx;
                if (yy >= 0 && yy < H && x2 >= 0 && x2 < W)
                    acc += x[c * HW + yy * W + x2] * w1[((k * 3 + c) * 3 + (dy + 1)) * 3 + (dx + 1)];
            }
    hP[(size_t)((y + 1) * PW + xx + 1) * 64 + k] = (_Float16)(acc > 0.f ? acc : 0.f);
}

// ---------------- enc2: conv 64->128 + relu via WMMA ----------------
// padded NHWC f16 in -> NCHW f32 out.  One wave per (x-tile, row, n-tile).
__global__ void enc2_wmma(const _Float16* __restrict__ hP, const _Float16* __restrict__ wp,
                          const float* __restrict__ bias, float* __restrict__ u) {
    int l = threadIdx.x;            // one wave
    int m = l & 15, g = l >> 4;
    int x0 = blockIdx.x * 16;
    int y  = blockIdx.y;
    int n0 = blockIdx.z * 16;
    // per-lane base pointers; all tap/chunk offsets are compile-time constants
    const _Float16* abase = hP + (size_t)((y + 1) * PW + x0 + m + 1) * 64 + g * 8;
    const _Float16* bbase = wp + (size_t)(n0 + m) * 64 + g * 8;
    vf8 acc = {};
    #pragma unroll
    for (int tap = 0; tap < 9; ++tap) {
        const int dy = tap / 3 - 1, dx = tap % 3 - 1;
        const int aoff = (dy * PW + dx) * 64;
        #pragma unroll
        for (int c0 = 0; c0 < 64; c0 += 32) {
            vh16 a = CAT16(*(const vh8*)(abase + aoff + c0),
                           *(const vh8*)(abase + aoff + c0 + 16));
            vh16 b = CAT16(*(const vh8*)(bbase + tap * 8192 + c0),
                           *(const vh8*)(bbase + tap * 8192 + c0 + 16));
            acc = __builtin_amdgcn_wmma_f32_16x16x32_f16(false, a, false, b,
                                                         (short)0, acc, false, false);
        }
    }
    float bk = bias[n0 + m];
    float4 lo, hi;
    lo.x = fmaxf(acc[0] + bk, 0.f); lo.y = fmaxf(acc[1] + bk, 0.f);
    lo.z = fmaxf(acc[2] + bk, 0.f); lo.w = fmaxf(acc[3] + bk, 0.f);
    hi.x = fmaxf(acc[4] + bk, 0.f); hi.y = fmaxf(acc[5] + bk, 0.f);
    hi.z = fmaxf(acc[6] + bk, 0.f); hi.w = fmaxf(acc[7] + bk, 0.f);
    float* op = u + (size_t)(n0 + m) * HW + y * W + x0 + 8 * g;
    *(float4*)(op)     = lo;
    *(float4*)(op + 4) = hi;
}

// ---------------- fused curvature diffusion step (fp32, one pass over u) ----------------
__global__ void diffuse_step(const float* __restrict__ uin, float* __restrict__ uout) {
    __shared__ float su[20][21];
    __shared__ float sux[18][19];
    __shared__ float suy[18][19];
    int tid = threadIdx.x;                 // 256 threads = 16x16 tile
    int c = blockIdx.z;
    int bx = blockIdx.x * 16 - 2, by = blockIdx.y * 16 - 2;
    const float* up = uin + (size_t)c * HW;
    for (int i = tid; i < 400; i += 256) {
        int r = i / 20, col = i % 20;
        int gy = by + r, gx = bx + col;
        float v = 0.f;
        if (gy >= 0 && gy < H && gx >= 0 && gx < W) v = up[gy * W + gx];
        su[r][col] = v;
    }
    __syncthreads();
    for (int i = tid; i < 324; i += 256) {
        int r = i / 18, col = i % 18;      // center at su[r+1][col+1]
        int gy = by + r + 1, gx = bx + col + 1;
        float vx = 0.f, vy = 0.f;
        if (gy >= 0 && gy < H && gx >= 0 && gx < W) {
            float a00 = su[r][col],   a01 = su[r][col+1],   a02 = su[r][col+2];
            float a10 = su[r+1][col],                       a12 = su[r+1][col+2];
            float a20 = su[r+2][col], a21 = su[r+2][col+1], a22 = su[r+2][col+2];
            vx = (a02 + 2.f*a12 + a22) - (a00 + 2.f*a10 + a20);   // SOBEL
            vy = (a20 + 2.f*a21 + a22) - (a00 + 2.f*a01 + a02);   // SOBEL^T
        }
        sux[r][col] = vx; suy[r][col] = vy;
    }
    __syncthreads();
    int r = tid >> 4, col = tid & 15;      // center: su[r+2][col+2], sux/suy[r+1][col+1]
    float x00 = sux[r][col],   x01 = sux[r][col+1],   x02 = sux[r][col+2];
    float x10 = sux[r+1][col],                        x12 = sux[r+1][col+2];
    float x20 = sux[r+2][col], x21 = sux[r+2][col+1], x22 = sux[r+2][col+2];
    float uxx = (x02 + 2.f*x12 + x22) - (x00 + 2.f*x10 + x20);
    float uxy = (x20 + 2.f*x21 + x22) - (x00 + 2.f*x01 + x02);
    float y00 = suy[r][col],   y01 = suy[r][col+1],   y02 = suy[r][col+2];
    float y20 = suy[r+2][col], y21 = suy[r+2][col+1], y22 = suy[r+2][col+2];
    float uyy = (y20 + 2.f*y21 + y22) - (y00 + 2.f*y01 + y02);
    float ux = sux[r+1][col+1], uy = suy[r+1][col+1];
    float num = uxx * uyy - uxy * uxy;
    float t = 1.f + ux * ux + uy * uy;
    float res = num / (t * t + 1e-6f);
    int gy = blockIdx.y * 16 + r, gx = blockIdx.x * 16 + col;
    uout[(size_t)c * HW + gy * W + gx] = su[r + 2][col + 2] + 0.1f * res;
}

// ---------------- NCHW f32 -> PADDED NHWC f16 for dec1 WMMA ----------------
__global__ void u_to_nhwc(const float* __restrict__ u, _Float16* __restrict__ uP) {
    int idx = blockIdx.x * 256 + threadIdx.x;        // HW*128
    if (idx >= HW * 128) return;
    int c = idx & 127; int p = idx >> 7;
    int y = p / W, x = p % W;
    uP[(size_t)((y + 1) * PW + x + 1) * 128 + c] = (_Float16)u[(size_t)c * HW + p];
}

// ---------------- dec1: conv 128->64 + relu via WMMA, padded NHWC f16 -> NHWC f16 ----------------
__global__ void dec1_wmma(const _Float16* __restrict__ uP, const _Float16* __restrict__ wp,
                          const float* __restrict__ bias, _Float16* __restrict__ d) {
    int l = threadIdx.x;
    int m = l & 15, g = l >> 4;
    int x0 = blockIdx.x * 16;
    int y  = blockIdx.y;
    int n0 = blockIdx.z * 16;
    const _Float16* abase = uP + (size_t)((y + 1) * PW + x0 + m + 1) * 128 + g * 8;
    const _Float16* bbase = wp + (size_t)(n0 + m) * 128 + g * 8;
    vf8 acc = {};
    #pragma unroll
    for (int tap = 0; tap < 9; ++tap) {
        const int dy = tap / 3 - 1, dx = tap % 3 - 1;
        const int aoff = (dy * PW + dx) * 128;
        #pragma unroll
        for (int c0 = 0; c0 < 128; c0 += 32) {
            vh16 a = CAT16(*(const vh8*)(abase + aoff + c0),
                           *(const vh8*)(abase + aoff + c0 + 16));
            vh16 b = CAT16(*(const vh8*)(bbase + tap * 8192 + c0),
                           *(const vh8*)(bbase + tap * 8192 + c0 + 16));
            acc = __builtin_amdgcn_wmma_f32_16x16x32_f16(false, a, false, b,
                                                         (short)0, acc, false, false);
        }
    }
    float bk = bias[n0 + m];
    _Float16* op = d + (size_t)(y * W + x0 + 8 * g) * 64 + n0 + m;
    #pragma unroll
    for (int r = 0; r < 8; ++r) {
        float v = acc[r] + bk;
        v = v > 0.f ? v : 0.f;
        op[(size_t)r * 64] = (_Float16)v;
    }
}

// ---------------- dec2: conv 64->3 (no relu), NHWC f16 -> NCHW f32 output ----------------
__global__ void dec2_kernel(const _Float16* __restrict__ d, const float* __restrict__ w,
                            const float* __restrict__ b, float* __restrict__ out) {
    int p = blockIdx.x * 256 + threadIdx.x;
    if (p >= HW) return;
    int y = p / W, x = p % W;
    float a0 = b[0], a1 = b[1], a2 = b[2];
    #pragma unroll
    for (int dy = -1; dy <= 1; ++dy)
        #pragma unroll
        for (int dx = -1; dx <= 1; ++dx) {
            int yy = y + dy, xx = x + dx;
            if (yy < 0 || yy >= H || xx < 0 || xx >= W) continue;
            int t = (dy + 1) * 3 + (dx + 1);
            const _Float16* dp = d + (size_t)(yy * W + xx) * 64;
            for (int c = 0; c < 64; ++c) {
                float v = (float)dp[c];
                a0 += v * w[(0 * 64 + c) * 9 + t];   // dec_w2 [3,64,3,3]
                a1 += v * w[(1 * 64 + c) * 9 + t];
                a2 += v * w[(2 * 64 + c) * 9 + t];
            }
        }
    out[0 * HW + p] = a0;
    out[1 * HW + p] = a1;
    out[2 * HW + p] = a2;
}

extern "C" void kernel_launch(void* const* d_in, const int* in_sizes, int n_in,
                              void* d_out, int out_size, void* d_ws, size_t ws_size,
                              hipStream_t stream) {
    const float* x      = (const float*)d_in[0];
    const float* enc_w1 = (const float*)d_in[1];
    const float* enc_b1 = (const float*)d_in[2];
    const float* enc_w2 = (const float*)d_in[3];
    const float* enc_b2 = (const float*)d_in[4];
    const float* dec_w1 = (const float*)d_in[5];
    const float* dec_b1 = (const float*)d_in[6];
    const float* dec_w2 = (const float*)d_in[7];
    const float* dec_b2 = (const float*)d_in[8];
    float* out = (float*)d_out;

    char* ws = (char*)d_ws;
    size_t oUA = 0;                                    // u ping  [128,192,192] f32
    size_t oUB = oUA + (size_t)128 * HW * 4;           // u pong
    size_t oH  = oUB + (size_t)128 * HW * 4;           // h padded NHWC [194,194,64] f16
    size_t oUN = oH  + (size_t)64 * PW * PW * 2;       // u padded NHWC [194,194,128] f16
    size_t oDN = oUN + (size_t)128 * PW * PW * 2;      // d NHWC [192,192,64] f16
    size_t oW2 = oDN + (size_t)64 * HW * 2;            // packed enc_w2 f16 (9*128*64)
    size_t oWD = oW2 + (size_t)9 * 128 * 64 * 2;       // packed dec_w1 f16 (9*64*128)

    float*    uA  = (float*)(ws + oUA);
    float*    uB  = (float*)(ws + oUB);
    _Float16* hP  = (_Float16*)(ws + oH);
    _Float16* uP  = (_Float16*)(ws + oUN);
    _Float16* dN  = (_Float16*)(ws + oDN);
    _Float16* w2p = (_Float16*)(ws + oW2);
    _Float16* wdp = (_Float16*)(ws + oWD);

    // zero padded buffers (halo stays 0; interiors overwritten by producers)
    int nH32 = (64 * PW * PW * 2) / 4;
    int nU32 = (128 * PW * PW * 2) / 4;
    zero_fill<<<(nH32 + 255) / 256, 256, 0, stream>>>((uint32_t*)hP, nH32);
    zero_fill<<<(nU32 + 255) / 256, 256, 0, stream>>>((uint32_t*)uP, nU32);

    pack_w_enc2<<<(9 * 128 * 64 + 255) / 256, 256, 0, stream>>>(enc_w2, w2p);
    pack_w_dec1<<<(9 * 64 * 128 + 255) / 256, 256, 0, stream>>>(dec_w1, wdp);

    enc1_kernel<<<(HW * 64 + 255) / 256, 256, 0, stream>>>(x, enc_w1, enc_b1, hP);

    enc2_wmma<<<dim3(12, 192, 8), 32, 0, stream>>>(hP, w2p, enc_b2, uA);

    for (int s = 0; s < 100; ++s) {
        const float* in = (s & 1) ? uB : uA;
        float* o        = (s & 1) ? uA : uB;
        diffuse_step<<<dim3(12, 12, 128), 256, 0, stream>>>(in, o);
    }
    // 100 steps: ends in uA

    u_to_nhwc<<<(HW * 128 + 255) / 256, 256, 0, stream>>>(uA, uP);

    dec1_wmma<<<dim3(12, 192, 4), 32, 0, stream>>>(uP, wdp, dec_b1, dN);

    dec2_kernel<<<(HW + 255) / 256, 256, 0, stream>>>(dN, dec_w2, dec_b2, out);
}
